// ProbProtoSegHead_44478681317595
// MI455X (gfx1250) — matrix operand order repaired and checked by reading, not compiled, not run on hardware
//
#include <hip/hip_runtime.h>

// Wasserstein-2 similarity head, fused dual-GEMM with bf16 WMMA.
//   out[n,cm] = 2*(x.pm + lam*sqrt(xv).sqrt(pv))[n,cm] - 2 - lam*(sum xv[n] + sum pv[cm])
// Prototypes pre-converted once into WMMA B-fragment layout [ks][tile][frag][512] in d_ws:
//   frag0/1 = bf16 hi/lo split of proto_mean, frag2 = lam*sqrt(proto_var) (lam=2^-8, exact).
// Main kernel: wave = 16 rows x ALL 12 column tiles (A converted once, reused 12x);
// each block stages the per-k-step 36KB B slab into LDS once (double-buffered) so the
// 8 waves read B from LDS instead of 8x from L2 -> HBM (x, x_var, out) stays the roof.

typedef __attribute__((ext_vector_type(16))) __bf16 v16bf;
typedef __attribute__((ext_vector_type(8)))  float  v8f;
typedef __attribute__((ext_vector_type(4)))  float  v4f;
typedef __attribute__((ext_vector_type(4)))  unsigned int v4u;

#define NN   131072
#define DD   256
#define CM   190                  // C*M = 19*10
#define NT   12                   // column tiles (padded to 192)
#define KS   8                    // k-steps of 32
#define LAM  (1.0f / 256.0f)

#define FRAGB   512               // bf16 elements per fragment (32 lanes x 16)
#define STEP_E  (NT * 3 * FRAGB)  // 18432 bf16 per k-step slab (36 KB)
#define B_ELEMS (KS * STEP_E)     // 147456 bf16 total (288 KB)
#define NCHUNK  9                 // 36864 B / (256 thr * 16 B)

// ---------- prep 1: build B slabs (96 blocks x 32 lanes) ----------
// Fragment (ks,t): lane l -> col t*16+(l&15) (clamped), K = ks*32 + (l>>4)*16 + 0..15
__global__ __launch_bounds__(32) void
prep_frags(const float* __restrict__ pm, const float* __restrict__ pv,
           __bf16* __restrict__ B)
{
    const int b    = blockIdx.x;            // t*8 + ks
    const int lane = threadIdx.x;
    const int t = b >> 3, ks = b & 7;
    const int l15 = lane & 15, half = lane >> 4;
    int col = t * 16 + l15; if (col > CM - 1) col = CM - 1;
    const int kbase = ks * 32 + half * 16;
    const float* pmrow = pm + (size_t)col * DD + kbase;
    const float* pvrow = pv + (size_t)col * DD + kbase;
    const size_t o = ((size_t)(ks * NT + t) * 3) * FRAGB + (size_t)lane * 16;
    #pragma unroll
    for (int i = 0; i < 16; ++i) {
        float v = pmrow[i];
        __bf16 h = (__bf16)v;
        B[o + i]             = h;                                  // hi
        B[o + FRAGB + i]     = (__bf16)(v - (float)h);             // lo
        B[o + 2 * FRAGB + i] = (__bf16)(LAM * __builtin_amdgcn_sqrtf(pvrow[i]));
    }
}

// ---------- prep 2: proto_var row sums (190 blocks x 32 lanes) ----------
__global__ __launch_bounds__(32) void
prep_pvs(const float* __restrict__ pv, float* __restrict__ pvs)
{
    const int row = blockIdx.x, lane = threadIdx.x;
    const float* p = pv + (size_t)row * DD + lane * 8;
    float s = 0.0f;
    #pragma unroll
    for (int i = 0; i < 8; ++i) s += p[i];
    #pragma unroll
    for (int m = 16; m >= 1; m >>= 1) s += __shfl_xor(s, m, 32);
    if (lane == 0) pvs[row] = s;
}

// ---------- main: 1024 blocks x 256 threads; wave = 16 rows x 12 col tiles ----------
__global__ __launch_bounds__(256) void
wass_main(const float* __restrict__ x, const float* __restrict__ x_var,
          const __bf16* __restrict__ Bg, const float* __restrict__ pvs,
          float* __restrict__ out)
{
    __shared__ __bf16 smem[2 * STEP_E];            // 72 KB double buffer (of 320 KB/WGP)

    const int tid  = threadIdx.x;
    const int lane = tid & 31;
    const int wave = tid >> 5;
    const int half = lane >> 4, l15 = lane & 15;
    const int n0 = (blockIdx.x * 8 + wave) * 16;

    // A k order used here (contiguous, same as B frags): lane<16 -> row l15, K {0..15}; lane>=16 -> {16..31}
    const float* xrow  = x     + (size_t)(n0 + l15) * DD + half * 16;
    const float* xvrow = x_var + (size_t)(n0 + l15) * DD + half * 16;
    const size_t lb = (size_t)lane * 16;           // lane offset inside a fragment

    v8f acc[NT];
    #pragma unroll
    for (int t = 0; t < NT; ++t) acc[t] = (v8f){};
    float xv_sum = 0.0f;

    // prologue: stage slab 0 into buffer 0 (contiguous 36 KB cooperative copy)
    #pragma unroll
    for (int j = 0; j < NCHUNK; ++j) {
        v4u c = *(const v4u*)((const char*)Bg + (size_t)(tid + j * 256) * 16);
        *(v4u*)((char*)smem + (size_t)(tid + j * 256) * 16) = c;
    }
    __syncthreads();

    for (int ks = 0; ks < KS; ++ks) {
        const int cur = ks & 1;
        const __bf16* sb = smem + cur * STEP_E;

        // prefetch next B slab into registers (hides global latency behind WMMAs)
        v4u g[NCHUNK];
        if (ks < KS - 1) {
            #pragma unroll
            for (int j = 0; j < NCHUNK; ++j)
                g[j] = *(const v4u*)((const char*)Bg + (size_t)(ks + 1) * (STEP_E * 2)
                                                     + (size_t)(tid + j * 256) * 16);
        }

        const int k = ks * 32;
        float a[16], av[16];
        *(v4f*)(a  + 0)  = *(const v4f*)(xrow  + k + 0);
        *(v4f*)(a  + 4)  = *(const v4f*)(xrow  + k + 4);
        *(v4f*)(a  + 8)  = *(const v4f*)(xrow  + k + 8);
        *(v4f*)(a  + 12) = *(const v4f*)(xrow  + k + 12);
        *(v4f*)(av + 0)  = *(const v4f*)(xvrow + k + 0);
        *(v4f*)(av + 4)  = *(const v4f*)(xvrow + k + 4);
        *(v4f*)(av + 8)  = *(const v4f*)(xvrow + k + 8);
        *(v4f*)(av + 12) = *(const v4f*)(xvrow + k + 12);

        v16bf a_hi, a_lo, sa;
        #pragma unroll
        for (int i = 0; i < 16; ++i) {
            __bf16 ah = (__bf16)a[i];
            a_hi[i] = ah;
            a_lo[i] = (__bf16)(a[i] - (float)ah);               // hi/lo split: fp32-class dot
            sa[i]   = (__bf16)__builtin_amdgcn_sqrtf(av[i]);    // single v_sqrt_f32
            xv_sum += av[i];                                    // row sum of x_var
        }

        // A fragments converted ONCE, reused across all 12 column tiles; B from LDS.
        #pragma unroll
        for (int t = 0; t < NT; ++t) {
            const __bf16* fb = sb + (size_t)t * 3 * FRAGB + lb;
            v16bf bh = *(const v16bf*)(fb);
            v16bf bl = *(const v16bf*)(fb + FRAGB);
            v16bf bq = *(const v16bf*)(fb + 2 * FRAGB);
            acc[t] = __builtin_amdgcn_wmma_f32_16x16x32_bf16(false, a_hi, false, bh,
                                                             (short)0, acc[t], false, false);
            acc[t] = __builtin_amdgcn_wmma_f32_16x16x32_bf16(false, a_hi, false, bl,
                                                             (short)0, acc[t], false, false);
            acc[t] = __builtin_amdgcn_wmma_f32_16x16x32_bf16(false, a_lo, false, bh,
                                                             (short)0, acc[t], false, false);
            acc[t] = __builtin_amdgcn_wmma_f32_16x16x32_bf16(false, sa,   false, bq,
                                                             (short)0, acc[t], false, false);
        }

        // write next slab into the other buffer; one barrier per iteration suffices
        // (buffer written here was last *read* one iteration ago, fenced by that barrier)
        if (ks < KS - 1) {
            char* dst = (char*)smem + (1 - cur) * (STEP_E * 2);
            #pragma unroll
            for (int j = 0; j < NCHUNK; ++j)
                *(v4u*)(dst + (size_t)(tid + j * 256) * 16) = g[j];
        }
        __syncthreads();
    }

    // lane L and L^16 jointly cover a full row of x_var
    xv_sum += __shfl_xor(xv_sum, 16, 32);   // lane L holds sum(x_var[n0 + (L&15)])

    // per-row bias, aligned to C layout (VGPR v -> row v + 8*half)
    float xvb[8];
    #pragma unroll
    for (int v = 0; v < 8; ++v)
        xvb[v] = LAM * __shfl(xv_sum, v + 8 * half, 32);

    #pragma unroll
    for (int t = 0; t < NT; ++t) {
        const int col = t * 16 + l15;
        const int colc = col > CM - 1 ? CM - 1 : col;
        const float pvb = LAM * pvs[colc];
        const bool valid = col < CM;
        #pragma unroll
        for (int v = 0; v < 8; ++v) {
            float r = 2.0f * acc[t][v] - 2.0f - xvb[v] - pvb;
            if (valid) {
                const int n = n0 + v + 8 * half;
                out[(size_t)n * CM + col] = r;
            }
        }
    }
}

extern "C" void kernel_launch(void* const* d_in, const int* in_sizes, int n_in,
                              void* d_out, int out_size, void* d_ws, size_t ws_size,
                              hipStream_t stream) {
    const float* x     = (const float*)d_in[0];   // [N, D]
    const float* x_var = (const float*)d_in[1];   // [N, D]
    const float* pmean = (const float*)d_in[2];   // [C, M, D] -> [CM, D]
    const float* pvar  = (const float*)d_in[3];   // [C, M, D] -> [CM, D]
    float* out = (float*)d_out;                   // [N, CM]
    (void)in_sizes; (void)n_in; (void)out_size; (void)ws_size;

    // ws layout: B slabs (B_ELEMS bf16, 288 KB) | pvs (190 f32)
    __bf16* B   = (__bf16*)d_ws;
    float*  pvs = (float*)(B + B_ELEMS);

    prep_frags<<<dim3(NT * KS), dim3(32), 0, stream>>>(pmean, pvar, B);
    prep_pvs  <<<dim3(CM),      dim3(32), 0, stream>>>(pvar, pvs);
    wass_main <<<dim3(NN / (16 * 8)), dim3(256), 0, stream>>>(x, x_var, B, pvs, out);
}
